// DeepLstm_79328045957854
// MI455X (gfx1250) — matrix operand order repaired and checked by reading, not compiled, not run on hardware
//
#include <hip/hip_runtime.h>
#include <hip/hip_bf16.h>

typedef __bf16 bf16;
typedef __attribute__((ext_vector_type(16))) __bf16 v16bf;
typedef __attribute__((ext_vector_type(8)))  __bf16 v8bf;
typedef __attribute__((ext_vector_type(8)))  float  v8f;
typedef unsigned int v4u __attribute__((ext_vector_type(4)));
typedef int          v8i __attribute__((ext_vector_type(8)));
typedef int          v4i __attribute__((ext_vector_type(4)));

#define NB   64      // batch
#define TTOT 512     // timesteps
#define FF   384     // input features
#define HH   384     // hidden
#define GG   1536    // 4*H gates
#define NL   3       // layers
#define TCH  128     // time chunk for pregate buffer
#define BH   (NB*HH)

#if defined(__gfx1250__) && __has_builtin(__builtin_amdgcn_tensor_load_to_lds)
#define USE_TDM 1
#else
#define USE_TDM 0
#endif

union FragBF { v16bf v; v8bf h[2]; };

__device__ __forceinline__ float sigmoidf_(float x) { return 1.0f / (1.0f + __expf(-x)); }

#if USE_TDM
// ---------------------------------------------------------------------------
// Tensor Data Mover: copy a [rows x 64B] tile (global row stride 768B) into
// LDS with 80B row stride (64B data + 16B pad via D# pad fields).
//   data_size = 8B (code 3); tile_dim0 = 8 units; tensor_dim0_stride = 96 units
//   pad_interval code 3 -> pad after 64B; pad_amount code 3 -> 16B (4 DWORDs)
// Issued by one wave; EXEC ignored by TDM; tracked with TENSORcnt.
// This toolchain's builtin is the 6-arg form:
//   (uint32x4 g0, int32x8 g1, int32x4 g2, int32x4 g3, int32x8 gx, i32 cpol)
// ---------------------------------------------------------------------------
__device__ __forceinline__ void tdm_load_rows64B(unsigned lds_off, const void* gptr, int rows)
{
    unsigned long long ga = (unsigned long long)(size_t)gptr;
    v4u g0;
    g0[0] = 1u;                                            // count=1 (valid user D#)
    g0[1] = lds_off;                                       // lds_addr (bytes)
    g0[2] = (unsigned)ga;                                  // global_addr[31:0]
    g0[3] = (unsigned)((ga >> 32) & 0x01FFFFFFu)           // global_addr[56:32]
          | 0x80000000u;                                   // type = 2 ("image")
    v8i g1;
    g1[0] = (3 << 16)            // data_size = 8B
          | (1 << 20)            // pad_enable
          | (3 << 22)            // pad_interval: pad after 16 DWORDs (64B)
          | (3 << 25);           // pad_amount: 4 DWORDs (16B)
    g1[1] = (96 & 0xFFFF) << 16;                 // tensor_dim0[15:0] = 96 units
    g1[2] = (96 >> 16) & 0xFFFF;                 // tensor_dim0[31:16] | tensor_dim1[15:0]=0
    g1[3] = (int)((1u << 20) >> 16)              // tensor_dim1[31:16]  (dim1 = 1M rows)
          | (8 << 16);                           // tile_dim0 = 8 units (64B)
    g1[4] = rows & 0xFFFF;                       // tile_dim1 = rows; tile_dim2 = 0
    g1[5] = 96;                                  // tensor_dim0_stride[31:0] = 96 units (768B)
    g1[6] = 0;                                   // stride0 hi | tensor_dim1_stride lo
    g1[7] = 0;
    v4i gz4 = {0, 0, 0, 0};                      // groups 2/3 unused (2D tensor)
    v8i gz8 = {0, 0, 0, 0, 0, 0, 0, 0};
    __builtin_amdgcn_tensor_load_to_lds(g0, g1, gz4, gz4, gz8, 0);
}
#endif

// ---------------------------------------------------------------------------
// Kernel A: pregate = X(bf16)[M,384] @ W(bf16)[384,1536] + bias, fp32 out.
// 128x128 tile per 256-thread block (8 waves), each wave 2 Mtiles x 4 Ntiles.
// A tile staged by TDM (when available); B tile staged transposed by threads.
// ---------------------------------------------------------------------------
__global__ __launch_bounds__(256)
void k_pregate_gemm(const bf16* __restrict__ X, const bf16* __restrict__ W,
                    const float* __restrict__ bias, float* __restrict__ out)
{
    __shared__ bf16 lA[128 * 40];   // row-major [row][k], 80B stride (64B + 16B pad)
    __shared__ bf16 lB[128 * 40];   // col-major [col][k]

    const int tid   = threadIdx.x;
    const int lane  = tid & 31;
    const int wave  = tid >> 5;
    const int rbase = blockIdx.y * 128;
    const int cbase = blockIdx.x * 128;

    const int mt0 = (wave & 3) * 2;     // 2 M-tiles per wave
    const int nt0 = (wave >> 2) * 4;    // 4 N-tiles per wave

    v8f zero = {};
    v8f acc[2][4];
    #pragma unroll
    for (int i = 0; i < 2; ++i)
        #pragma unroll
        for (int j = 0; j < 4; ++j) acc[i][j] = zero;

    const int bk = tid >> 3, bc = tid & 7;   // B staging: k, 16-col group
#if !USE_TDM
    const int ar = tid >> 1, ah = tid & 1;   // A staging: row, 16-elem half
#endif

    for (int ks = 0; ks < FF; ks += 32) {
        // ---- A tile 128x32 ----
#if USE_TDM
        if (wave == 0)
            tdm_load_rows64B((unsigned)(size_t)(void*)lA,
                             X + (size_t)rbase * FF + ks, 128);
#else
        {
            const bf16* g = X + (size_t)(rbase + ar) * FF + ks + ah * 16;
            v8bf d0 = *(const v8bf*)g;
            v8bf d1 = *(const v8bf*)(g + 8);
            bf16* s = lA + ar * 40 + ah * 16;
            *(v8bf*)s       = d0;
            *(v8bf*)(s + 8) = d1;
        }
#endif
        // ---- B tile 32x128 -> LDS transposed [col][k] ----
        {
            const bf16* g = W + (size_t)(ks + bk) * GG + cbase + bc * 16;
            v8bf d0 = *(const v8bf*)g;
            v8bf d1 = *(const v8bf*)(g + 8);
            #pragma unroll
            for (int j = 0; j < 8; ++j) lB[(bc * 16 + j) * 40 + bk]     = d0[j];
            #pragma unroll
            for (int j = 0; j < 8; ++j) lB[(bc * 16 + 8 + j) * 40 + bk] = d1[j];
        }
#if USE_TDM
        if (wave == 0) __builtin_amdgcn_s_wait_tensorcnt(0);
#endif
        __syncthreads();

        const int selA = (lane >= 16) ? 16 : 0;   // A: lanes>=16 take K 8..15 / 24..31
        const int selB = (lane >= 16) ? 16 : 0;   // B: lanes>=16 take K 8..15 / 24..31
        FragBF af[2], bfr[4];
        #pragma unroll
        for (int i = 0; i < 2; ++i) {
            const char* p = (const char*)lA + ((mt0 + i) * 16 + (lane & 15)) * 80 + selA;
            af[i].h[0] = *(const v8bf*)p;
            af[i].h[1] = *(const v8bf*)(p + 32);
        }
        #pragma unroll
        for (int j = 0; j < 4; ++j) {
            const char* p = (const char*)lB + ((nt0 + j) * 16 + (lane & 15)) * 80 + selB;
            bfr[j].h[0] = *(const v8bf*)p;
            bfr[j].h[1] = *(const v8bf*)(p + 32);
        }
        #pragma unroll
        for (int i = 0; i < 2; ++i)
            #pragma unroll
            for (int j = 0; j < 4; ++j)
                acc[i][j] = __builtin_amdgcn_wmma_f32_16x16x32_bf16(
                    false, af[i].v, false, bfr[j].v, (short)0, acc[i][j], false, false);
        __syncthreads();
    }

    const int rowSel = (lane >= 16) ? 8 : 0;
    #pragma unroll
    for (int j = 0; j < 4; ++j) {
        const int n = cbase + (nt0 + j) * 16 + (lane & 15);
        const float bb = bias[n];
        #pragma unroll
        for (int i = 0; i < 2; ++i) {
            const int mrow = rbase + (mt0 + i) * 16 + rowSel;
            #pragma unroll
            for (int r = 0; r < 8; ++r)
                out[(size_t)(mrow + r) * GG + n] = acc[i][j][r] + bb;
        }
    }
}

// ---------------------------------------------------------------------------
// Kernel B: one LSTM timestep. z = pregate[t] + hprev@U ; gates in registers.
// 12 blocks; each owns 32 h-columns across all 4 gates. Wave = (Mtile, hstrip);
// its 4 accumulators are i,f,g,o for the SAME (b,h) -> elementwise in VGPRs.
// ---------------------------------------------------------------------------
__global__ __launch_bounds__(256)
void k_lstm_step(const bf16* __restrict__ hprev, const bf16* __restrict__ U,
                 const float* __restrict__ pre, const float* __restrict__ cIn,
                 float* __restrict__ cOut, bf16* __restrict__ ysOut,
                 float* __restrict__ hOut)
{
    __shared__ bf16 lA[64 * 40];
    __shared__ bf16 lB[128 * 40];

    const int tid   = threadIdx.x;
    const int lane  = tid & 31;
    const int wave  = tid >> 5;
    const int hbase = blockIdx.x * 32;

    const int mt     = wave & 3;
    const int hstrip = wave >> 2;

    // warm L2/L0 for the epilogue's pregate rows while the GEMM runs
    __builtin_prefetch(pre + (size_t)(mt * 16) * GG + hbase, 0, 3);

    v8f zero = {};
    v8f acc[4];
    #pragma unroll
    for (int g = 0; g < 4; ++g) acc[g] = zero;

    const int bk = tid >> 3, bc = tid & 7;   // B: k 0..31, 16-col group
#if !USE_TDM
    const int ar = tid >> 2, aq = tid & 3;   // A: row 0..63, 8-elem quarter
#endif

    for (int ks = 0; ks < HH; ks += 32) {
        // ---- A tile 64x32 ----
#if USE_TDM
        if (wave == 0)
            tdm_load_rows64B((unsigned)(size_t)(void*)lA, hprev + ks, 64);
#else
        {
            const bf16* g = hprev + ar * HH + ks + aq * 8;
            *(v8bf*)(lA + ar * 40 + aq * 8) = *(const v8bf*)g;
        }
#endif
        // ---- B tile: 32 x 128 gate-scattered cols -> LDS [cidx][k] ----
        {
            const int gate = bc >> 1;
            const int colb = gate * HH + hbase + (bc & 1) * 16;
            const bf16* g = U + (size_t)(ks + bk) * GG + colb;
            v8bf d0 = *(const v8bf*)g;
            v8bf d1 = *(const v8bf*)(g + 8);
            #pragma unroll
            for (int j = 0; j < 8; ++j) lB[(bc * 16 + j) * 40 + bk]     = d0[j];
            #pragma unroll
            for (int j = 0; j < 8; ++j) lB[(bc * 16 + 8 + j) * 40 + bk] = d1[j];
        }
#if USE_TDM
        if (wave == 0) __builtin_amdgcn_s_wait_tensorcnt(0);
#endif
        __syncthreads();

        const int selA = (lane >= 16) ? 16 : 0;
        const int selB = (lane >= 16) ? 16 : 0;
        FragBF af;
        const char* pa = (const char*)lA + (mt * 16 + (lane & 15)) * 80 + selA;
        af.h[0] = *(const v8bf*)pa;
        af.h[1] = *(const v8bf*)(pa + 32);
        #pragma unroll
        for (int g = 0; g < 4; ++g) {
            FragBF bfr;
            const char* pb = (const char*)lB + ((g * 32 + hstrip * 16) + (lane & 15)) * 80 + selB;
            bfr.h[0] = *(const v8bf*)pb;
            bfr.h[1] = *(const v8bf*)(pb + 32);
            acc[g] = __builtin_amdgcn_wmma_f32_16x16x32_bf16(
                false, af.v, false, bfr.v, (short)0, acc[g], false, false);
        }
        __syncthreads();
    }

    const int h      = hbase + hstrip * 16 + (lane & 15);
    const int rowSel = (lane >= 16) ? 8 : 0;
    const int mbase  = mt * 16 + rowSel;
    #pragma unroll
    for (int r = 0; r < 8; ++r) {
        const int m = mbase + r;
        const float* pr = pre + (size_t)m * GG;
        float zi = sigmoidf_(acc[0][r] + pr[h]);
        float zf = sigmoidf_(acc[1][r] + pr[HH + h]);
        float zg = tanhf(acc[2][r] + pr[2 * HH + h]);
        float zo = sigmoidf_(acc[3][r] + pr[3 * HH + h]);
        const float cn = zf * cIn[m * HH + h] + zi * zg;
        const float hn = zo * tanhf(cn);
        cOut[m * HH + h]  = cn;
        hOut[m * HH + h]  = hn;
        ysOut[m * HH + h] = (bf16)hn;
    }
}

// ------------------------- small helper kernels ----------------------------
__global__ void k_cvt_x(const float* __restrict__ x, bf16* __restrict__ xs)
{   // [B,T,F] fp32 -> time-major [T,B,F] bf16
    int idx = blockIdx.x * 256 + threadIdx.x;
    int b = idx / (TTOT * FF);
    int rem = idx - b * (TTOT * FF);
    int t = rem / FF;
    int f = rem - t * FF;
    xs[(size_t)(t * NB + b) * FF + f] = (bf16)x[idx];
}

__global__ void k_cvt_wu(const float* __restrict__ W, const float* __restrict__ U,
                         bf16* __restrict__ Wb, bf16* __restrict__ Ub)
{
    int idx = blockIdx.x * 256 + threadIdx.x;
    Wb[idx] = (bf16)W[idx];
    Ub[idx] = (bf16)U[idx];
}

__global__ void k_init_layer(const float* __restrict__ h0, const float* __restrict__ c0,
                             bf16* __restrict__ hinit, float* __restrict__ c0buf)
{
    int idx = blockIdx.x * 256 + threadIdx.x;
    hinit[idx] = (bf16)h0[idx];
    c0buf[idx] = c0[idx];
}

__global__ void k_gather(const float* __restrict__ hF, const float* __restrict__ cbuf,
                         float* __restrict__ out)
{   // out = [last_output(B,H) | h_fin(3,B,H) | c_fin(3,B,H)]
    int idx = blockIdx.x * 256 + threadIdx.x;
    float v;
    if (idx < BH)                 v = hF[2 * BH + idx];
    else if (idx < BH + NL * BH)  v = hF[idx - BH];
    else {
        int j = idx - (BH + NL * BH);
        int l = j / BH;
        v = cbuf[(size_t)l * 2 * BH + (j - l * BH)];   // final parity = 0 (T even)
    }
    out[idx] = v;
}

// ---------------------------------------------------------------------------
extern "C" void kernel_launch(void* const* d_in, const int* in_sizes, int n_in,
                              void* d_out, int out_size, void* d_ws, size_t ws_size,
                              hipStream_t stream)
{
    const float* x  = (const float*)d_in[0];
    const float* W  = (const float*)d_in[1];
    const float* U  = (const float*)d_in[2];
    const float* b  = (const float*)d_in[3];
    const float* h0 = (const float*)d_in[4];
    const float* c0 = (const float*)d_in[5];
    float* out = (float*)d_out;

    char* ws = (char*)d_ws;
    size_t off = 0;
    auto alloc = [&](size_t bytes) -> char* {
        char* p = ws + off;
        off = (off + bytes + 255) & ~(size_t)255;
        return p;
    };
    bf16*  xsA     = (bf16*) alloc((size_t)TTOT * NB * FF * 2);
    bf16*  xsB     = (bf16*) alloc((size_t)TTOT * NB * FF * 2);
    bf16*  Wb      = (bf16*) alloc((size_t)NL * FF * GG * 2);
    bf16*  Ub      = (bf16*) alloc((size_t)NL * FF * GG * 2);
    float* pregate = (float*)alloc((size_t)TCH * NB * GG * 4);
    bf16*  hinit   = (bf16*) alloc((size_t)BH * 2);
    float* cbuf    = (float*)alloc((size_t)NL * 2 * BH * 4);
    float* hF      = (float*)alloc((size_t)NL * BH * 4);

    k_cvt_x <<<(NB * TTOT * FF) / 256, 256, 0, stream>>>(x, xsA);
    k_cvt_wu<<<(NL * FF * GG) / 256, 256, 0, stream>>>(W, U, Wb, Ub);

    const bf16* xin = xsA;
    bf16* yout = xsB;
    for (int l = 0; l < NL; ++l) {
        k_init_layer<<<BH / 256, 256, 0, stream>>>(h0 + l * BH, c0 + l * BH,
                                                   hinit, cbuf + (size_t)l * 2 * BH);
        for (int tc = 0; tc < TTOT / TCH; ++tc) {
            k_pregate_gemm<<<dim3(GG / 128, (TCH * NB) / 128), 256, 0, stream>>>(
                xin + (size_t)tc * TCH * NB * FF, Wb + (size_t)l * FF * GG,
                b + l * GG, pregate);
            for (int tt = 0; tt < TCH; ++tt) {
                int t = tc * TCH + tt;
                const bf16* aPrev = (t == 0) ? hinit : (yout + (size_t)(t - 1) * NB * HH);
                const float* cI = cbuf + (size_t)l * 2 * BH + (size_t)(t & 1) * BH;
                float*       cO = cbuf + (size_t)l * 2 * BH + (size_t)((t + 1) & 1) * BH;
                k_lstm_step<<<HH / 32, 256, 0, stream>>>(
                    aPrev, Ub + (size_t)l * FF * GG, pregate + (size_t)tt * NB * GG,
                    cI, cO, yout + (size_t)t * NB * HH, hF + (size_t)l * BH);
            }
        }
        const bf16* tmp = xin; xin = yout; yout = (bf16*)tmp;
    }
    k_gather<<<(BH + 2 * NL * BH) / 256, 256, 0, stream>>>(hF, cbuf, out);

    (void)n_in; (void)in_sizes; (void)out_size; (void)ws_size;
}
